// FunctionPredictor_68564857913749
// MI455X (gfx1250) — matrix operand order repaired and checked by reading, not compiled, not run on hardware
//
#include <hip/hip_runtime.h>
#include <hip/hip_bf16.h>

typedef __attribute__((ext_vector_type(16))) _Float16 v16h;
typedef __attribute__((ext_vector_type(8)))  float    v8f;

union FragH { v16h v; float4 q[2]; _Float16 h[16]; };
union H8    { _Float16 h[8]; float4 q; };

// ---------------------------------------------------------------------------
// Prep kernels
// ---------------------------------------------------------------------------

// xcat[r, 0:H] = seq, xcat[r, H:2H] = struct  (f32 -> f16)
__global__ __launch_bounds__(256) void build_xcat_kernel(
    const float* __restrict__ seq, const float* __restrict__ str,
    _Float16* __restrict__ xcat, int M, int H) {
  int idx = blockIdx.x * 256 + threadIdx.x;
  if (idx >= M * 2 * H) return;
  int r = idx / (2 * H), c = idx % (2 * H);
  float v = (c < H) ? seq[(size_t)r * H + c] : str[(size_t)r * H + (c - H)];
  xcat[idx] = (_Float16)v;
}

// Wt[n, k] = W[row_off + k, n]  (f32 -> f16, K padded with zeros to Kpad)
__global__ __launch_bounds__(256) void w_to_f16t_kernel(
    const float* __restrict__ W, _Float16* __restrict__ Wt,
    int Ksrc, int N, int Kpad, int row_off) {
  int idx = blockIdx.x * 256 + threadIdx.x;
  if (idx >= N * Kpad) return;
  int n = idx / Kpad, k = idx % Kpad;
  Wt[idx] = (k < Ksrc) ? (_Float16)W[(size_t)(row_off + k) * N + n] : (_Float16)0.f;
}

// cat[r, 0:320] = f, cat[r, 320:327] = ec (f16), cat[r, 327:352] = 0
__global__ __launch_bounds__(256) void build_cat_kernel(
    const _Float16* __restrict__ f, const float* __restrict__ ec,
    _Float16* __restrict__ cat, int M) {
  int idx = blockIdx.x * 256 + threadIdx.x;
  if (idx >= M * 352) return;
  int r = idx / 352, c = idx % 352;
  _Float16 v;
  if (c < 320)      v = f[(size_t)r * 320 + c];
  else if (c < 327) v = (_Float16)ec[(size_t)r * 7 + (c - 320)];
  else              v = (_Float16)0.f;
  cat[idx] = v;
}

__global__ __launch_bounds__(256) void softmax7_kernel(
    const float* __restrict__ elog, float* __restrict__ ec, int M) {
  int r = blockIdx.x * 256 + threadIdx.x;
  if (r >= M) return;
  float v[7]; float mx = -1e30f;
  #pragma unroll
  for (int c = 0; c < 7; ++c) { v[c] = elog[(size_t)r * 7 + c]; mx = fmaxf(mx, v[c]); }
  float s = 0.f;
  #pragma unroll
  for (int c = 0; c < 7; ++c) { v[c] = __expf(v[c] - mx); s += v[c]; }
  float inv = 1.f / s;
  #pragma unroll
  for (int c = 0; c < 7; ++c) ec[(size_t)r * 7 + c] = v[c] * inv;
}

// ---------------------------------------------------------------------------
// Generic 1-wave WMMA GEMM: out = act(A[M,K]f16 @ Bt[N,K]f16^T + bias)
// act: 0 = none, 1 = relu, 2 = sigmoid. outH (f16) and/or outF (f32).
// Requires: M % 16 == 0, K % 32 == 0. N arbitrary (guarded).
// ---------------------------------------------------------------------------
__global__ __launch_bounds__(32) void wmma_gemm_kernel(
    const _Float16* __restrict__ A, const _Float16* __restrict__ Bt,
    const float* __restrict__ bias, _Float16* __restrict__ outH,
    float* __restrict__ outF, int M, int N, int K, int act) {
  const int lane = threadIdx.x & 31;
  const int m0 = blockIdx.x * 16;
  const int n0 = blockIdx.y * 16;
  const int nc = n0 + (lane & 15);
  const int ncc = nc < N ? nc : N - 1;          // clamp: unused columns harmless
  float bv = bias ? bias[ncc] : 0.f;
  v8f acc;
  #pragma unroll
  for (int t = 0; t < 8; ++t) acc[t] = bv;
  // A fragment: lane&15 = row, halves 0..7 -> K = kb..kb+7, halves 8..15 -> K+16
  const _Float16* aBase = A + (size_t)(m0 + (lane & 15)) * K + ((lane >> 4) * 8);
  // B fragment: lane&15 = col, halves 0..7 -> K = kb..kb+7, halves 8..15 -> K+16
  const _Float16* bBase = Bt + (size_t)ncc * K + ((lane >> 4) * 8);
  for (int k0 = 0; k0 < K; k0 += 32) {
    if (k0 + 64 < K) __builtin_prefetch((const void*)(bBase + k0 + 64), 0, 3);
    FragH a, b;
    a.q[0] = *(const float4*)(aBase + k0);
    a.q[1] = *(const float4*)(aBase + k0 + 16);
    b.q[0] = *(const float4*)(bBase + k0);
    b.q[1] = *(const float4*)(bBase + k0 + 16);
    acc = __builtin_amdgcn_wmma_f32_16x16x32_f16(false, a.v, false, b.v,
                                                 (short)0, acc, false, false);
  }
  if (nc < N) {
    const int mb = m0 + (lane >> 4) * 8;        // C/D: vgpr t -> row mb+t, col nc
    #pragma unroll
    for (int t = 0; t < 8; ++t) {
      float v = acc[t];
      if (act == 1)      v = fmaxf(v, 0.f);
      else if (act == 2) v = 1.f / (1.f + __expf(-v));
      if (outH) outH[(size_t)(mb + t) * N + nc] = (_Float16)v;
      if (outF) outF[(size_t)(mb + t) * N + nc] = v;
    }
  }
}

// ---------------------------------------------------------------------------
// Fused PPI head, register-blocked over j.
// One wave per (i, 64-wide j block). Per wave:
//   hT[64,320] = relu(aP[i,:] + bP[j,:] + b1)   built in LDS (f16, 40 KB)
//   per c-tile (10): load all 10 K-fragments of W2t into registers (80 VGPRs),
//   reuse them across 4 j-tiles of A streamed from LDS  -> 4x less L2 B-traffic
//   fold @W3 per column, butterfly-reduce, sigmoid, store 64 pair scores.
// ---------------------------------------------------------------------------
#define PPI_JT 4
__global__ __launch_bounds__(32) void ppi_kernel(
    const _Float16* __restrict__ aP, const _Float16* __restrict__ bP,
    const float* __restrict__ b1, const float* __restrict__ b2,
    const _Float16* __restrict__ W2t, const float* __restrict__ W3,
    const float* __restrict__ b3p, float* __restrict__ out) {
  __shared__ _Float16 hT[PPI_JT * 16 * 320];    // 40 KB
  const int lane = threadIdx.x & 31;
  const int i = blockIdx.x;                     // pair row = b*256 + si
  const int j0 = blockIdx.y * (PPI_JT * 16);
  const int jTok0 = (i & ~255) + j0;            // batch base + j block
  // ---- build h tile: 64 rows x 320 cols, 2 lanes per row, 8 halves a step
  {
    const int kb = (lane & 1) * 160;
    const _Float16* arow = aP + (size_t)i * 320;
    for (int jj = (lane >> 1); jj < PPI_JT * 16; jj += 16) {
      const _Float16* brow = bP + (size_t)(jTok0 + jj) * 320;
      for (int k = kb; k < kb + 160; k += 8) {
        H8 av, bv, o;
        av.q = *(const float4*)(arow + k);
        bv.q = *(const float4*)(brow + k);
        const float4 bb0 = *(const float4*)(b1 + k);
        const float4 bb1 = *(const float4*)(b1 + k + 4);
        o.h[0] = (_Float16)fmaxf((float)av.h[0] + (float)bv.h[0] + bb0.x, 0.f);
        o.h[1] = (_Float16)fmaxf((float)av.h[1] + (float)bv.h[1] + bb0.y, 0.f);
        o.h[2] = (_Float16)fmaxf((float)av.h[2] + (float)bv.h[2] + bb0.z, 0.f);
        o.h[3] = (_Float16)fmaxf((float)av.h[3] + (float)bv.h[3] + bb0.w, 0.f);
        o.h[4] = (_Float16)fmaxf((float)av.h[4] + (float)bv.h[4] + bb1.x, 0.f);
        o.h[5] = (_Float16)fmaxf((float)av.h[5] + (float)bv.h[5] + bb1.y, 0.f);
        o.h[6] = (_Float16)fmaxf((float)av.h[6] + (float)bv.h[6] + bb1.z, 0.f);
        o.h[7] = (_Float16)fmaxf((float)av.h[7] + (float)bv.h[7] + bb1.w, 0.f);
        *(float4*)&hT[jj * 320 + k] = o.q;
      }
    }
  }
  __syncthreads();
  // ---- GEMM chain
  float pair[PPI_JT][8];
  #pragma unroll
  for (int jt = 0; jt < PPI_JT; ++jt)
    #pragma unroll
    for (int u = 0; u < 8; ++u) pair[jt][u] = 0.f;
  const _Float16* aLane = &hT[(lane & 15) * 320 + ((lane >> 4) * 8)];
  #pragma unroll 1
  for (int t = 0; t < 10; ++t) {                // c-tiles cover N=160
    const int c0 = t * 16;
    // load full K=320 B fragment set for this c-tile into registers
    const _Float16* bBase = W2t + (size_t)(c0 + (lane & 15)) * 320 + ((lane >> 4) * 8);
    FragH Bf[10];
    #pragma unroll
    for (int k = 0; k < 10; ++k) {
      Bf[k].q[0] = *(const float4*)(bBase + k * 32);
      Bf[k].q[1] = *(const float4*)(bBase + k * 32 + 16);
    }
    const float bcol = b2[c0 + (lane & 15)];
    const float w3 = W3[c0 + (lane & 15)];
    #pragma unroll
    for (int jt = 0; jt < PPI_JT; ++jt) {       // reuse Bf across 4 A tiles
      v8f acc;
      #pragma unroll
      for (int u = 0; u < 8; ++u) acc[u] = bcol;
      const _Float16* aB = aLane + (size_t)jt * 16 * 320;
      #pragma unroll
      for (int k = 0; k < 10; ++k) {            // A from LDS, 10 WMMAs
        FragH a;
        a.q[0] = *(const float4*)(aB + k * 32);
        a.q[1] = *(const float4*)(aB + k * 32 + 16);
        acc = __builtin_amdgcn_wmma_f32_16x16x32_f16(false, a.v, false, Bf[k].v,
                                                     (short)0, acc, false, false);
      }
      #pragma unroll
      for (int u = 0; u < 8; ++u) pair[jt][u] += fmaxf(acc[u], 0.f) * w3;
    }
  }
  // ---- reduce across the 16 columns held per half-wave, then sigmoid-store
  #pragma unroll
  for (int off = 1; off < 16; off <<= 1)
    #pragma unroll
    for (int jt = 0; jt < PPI_JT; ++jt)
      #pragma unroll
      for (int u = 0; u < 8; ++u)
        pair[jt][u] += __shfl_xor(pair[jt][u], off, 32);
  if ((lane & 15) == 0) {
    const float b3 = b3p[0];
    const int mb = (lane >> 4) * 8;             // lane0 -> j 0..7, lane16 -> 8..15
    #pragma unroll
    for (int jt = 0; jt < PPI_JT; ++jt)
      #pragma unroll
      for (int u = 0; u < 8; ++u)
        out[(size_t)i * 256 + j0 + jt * 16 + mb + u] =
            1.f / (1.f + __expf(-(pair[jt][u] + b3)));
  }
}

// out[r] = sigmoid(dot(A[r,:K], w) + bias), one wave per row
__global__ __launch_bounds__(128) void matvec_sigmoid_kernel(
    const _Float16* __restrict__ A, const float* __restrict__ w,
    const float* __restrict__ bias, float* __restrict__ out, int M, int K) {
  int wid = threadIdx.x >> 5, lane = threadIdx.x & 31;
  int r = blockIdx.x * 4 + wid;
  if (r >= M) return;
  float s = 0.f;
  for (int k = lane; k < K; k += 32) s += (float)A[(size_t)r * K + k] * w[k];
  #pragma unroll
  for (int off = 16; off; off >>= 1) s += __shfl_xor(s, off, 32);
  if (lane == 0) out[r] = 1.f / (1.f + __expf(-(s + bias[0])));
}

// ---------------------------------------------------------------------------
extern "C" void kernel_launch(void* const* d_in, const int* in_sizes, int n_in,
                              void* d_out, int out_size, void* d_ws, size_t ws_size,
                              hipStream_t stream) {
  (void)in_sizes; (void)n_in; (void)out_size; (void)ws_size;
  const int H = 320, G = 1000, M = 1024;        // B=4, S=256

  const float* seq     = (const float*)d_in[0];
  const float* str     = (const float*)d_in[1];
  const float* fus_W   = (const float*)d_in[2];
  const float* fus_b   = (const float*)d_in[3];
  const float* go_W1   = (const float*)d_in[4];
  const float* go_b1   = (const float*)d_in[5];
  const float* go_W2   = (const float*)d_in[6];
  const float* go_b2   = (const float*)d_in[7];
  const float* ppi_W1  = (const float*)d_in[8];
  const float* ppi_b1  = (const float*)d_in[9];
  const float* ppi_W2  = (const float*)d_in[10];
  const float* ppi_b2  = (const float*)d_in[11];
  const float* ppi_W3  = (const float*)d_in[12];
  const float* ppi_b3  = (const float*)d_in[13];
  const float* enz_W1  = (const float*)d_in[14];
  const float* enz_b1  = (const float*)d_in[15];
  const float* enz_W2  = (const float*)d_in[16];
  const float* enz_b2  = (const float*)d_in[17];
  const float* act_W1  = (const float*)d_in[18];
  const float* act_b1  = (const float*)d_in[19];
  const float* act_W2  = (const float*)d_in[20];
  const float* act_b2  = (const float*)d_in[21];
  const float* bind_W1 = (const float*)d_in[22];
  const float* bind_b1 = (const float*)d_in[23];
  const float* bind_W2 = (const float*)d_in[24];
  const float* bind_b2 = (const float*)d_in[25];
  const float* bind_W3 = (const float*)d_in[26];
  const float* bind_b3 = (const float*)d_in[27];

  // outputs (flat, return order): go, ppi, ec, act, bind
  float* outF     = (float*)d_out;
  float* go_out   = outF;
  float* ppi_out  = outF + 1024000;
  float* ec_out   = ppi_out + 262144;
  float* act_out  = ec_out + 7168;
  float* bind_out = act_out + 1024;

  // workspace bump allocator (256 B aligned)
  char* p = (char*)d_ws;
  auto alloc = [&](size_t bytes) -> void* {
    void* r = (void*)p; p += (bytes + 255) & ~size_t(255); return r;
  };
  _Float16* xcat    = (_Float16*)alloc((size_t)M * 640 * 2);
  _Float16* fusWt   = (_Float16*)alloc((size_t)320 * 640 * 2);
  _Float16* f_h     = (_Float16*)alloc((size_t)M * 320 * 2);
  _Float16* goW1t   = (_Float16*)alloc((size_t)640 * 320 * 2);
  _Float16* g1_h    = (_Float16*)alloc((size_t)M * 640 * 2);
  _Float16* goW2t   = (_Float16*)alloc((size_t)G * 640 * 2);
  _Float16* ppiW1aT = (_Float16*)alloc((size_t)320 * 320 * 2);
  _Float16* ppiW1bT = (_Float16*)alloc((size_t)320 * 320 * 2);
  _Float16* ppiW2t  = (_Float16*)alloc((size_t)160 * 320 * 2);
  _Float16* enzW1t  = (_Float16*)alloc((size_t)640 * 320 * 2);
  _Float16* e1_h    = (_Float16*)alloc((size_t)M * 640 * 2);
  _Float16* enzW2t  = (_Float16*)alloc((size_t)7 * 640 * 2);
  _Float16* cat_h   = (_Float16*)alloc((size_t)M * 352 * 2);
  _Float16* actW1t  = (_Float16*)alloc((size_t)320 * 352 * 2);
  _Float16* a1_h    = (_Float16*)alloc((size_t)M * 320 * 2);
  _Float16* bindW1t = (_Float16*)alloc((size_t)640 * 320 * 2);
  _Float16* bh1     = (_Float16*)alloc((size_t)M * 640 * 2);
  _Float16* bindW2t = (_Float16*)alloc((size_t)320 * 640 * 2);
  _Float16* bh2     = (_Float16*)alloc((size_t)M * 320 * 2);
  _Float16* aPh     = (_Float16*)alloc((size_t)M * 320 * 2);
  _Float16* bPh     = (_Float16*)alloc((size_t)M * 320 * 2);
  float*    elog    = (float*)alloc((size_t)M * 7 * 4);

  auto tpose = [&](const float* W, _Float16* Wt, int Ksrc, int N, int Kpad, int roff) {
    int total = N * Kpad;
    w_to_f16t_kernel<<<dim3((total + 255) / 256), dim3(256), 0, stream>>>(
        W, Wt, Ksrc, N, Kpad, roff);
  };
  auto gemm = [&](const _Float16* A, const _Float16* Bt, const float* bias,
                  _Float16* oH, float* oF, int Mm, int Nn, int Kk, int act) {
    wmma_gemm_kernel<<<dim3(Mm / 16, (Nn + 15) / 16), dim3(32), 0, stream>>>(
        A, Bt, bias, oH, oF, Mm, Nn, Kk, act);
  };

  // weight prep (f16, transposed/N-major; act_W1 K-padded 327 -> 352)
  tpose(fus_W,   fusWt,   640, 320,  640, 0);
  tpose(go_W1,   goW1t,   320, 640,  320, 0);
  tpose(go_W2,   goW2t,   640, G,    640, 0);
  tpose(ppi_W1,  ppiW1aT, 320, 320,  320, 0);
  tpose(ppi_W1,  ppiW1bT, 320, 320,  320, 320);
  tpose(ppi_W2,  ppiW2t,  320, 160,  320, 0);
  tpose(enz_W1,  enzW1t,  320, 640,  320, 0);
  tpose(enz_W2,  enzW2t,  640, 7,    640, 0);
  tpose(act_W1,  actW1t,  327, 320,  352, 0);
  tpose(bind_W1, bindW1t, 320, 640,  320, 0);
  tpose(bind_W2, bindW2t, 640, 320,  640, 0);

  // fusion: f = relu([seq|struct] @ fus_W + b)
  build_xcat_kernel<<<dim3((M * 640 + 255) / 256), dim3(256), 0, stream>>>(
      seq, str, xcat, M, H);
  gemm(xcat, fusWt, fus_b, f_h, nullptr, M, 320, 640, 1);

  // GO head: sigmoid(relu(f@W1+b1)@W2+b2) -> d_out
  gemm(f_h, goW1t, go_b1, g1_h, nullptr, M, 640, 320, 1);
  gemm(g1_h, goW2t, go_b2, nullptr, go_out, M, G, 640, 2);

  // PPI head: f16 projections then fused pairwise kernel
  gemm(f_h, ppiW1aT, nullptr, aPh, nullptr, M, 320, 320, 0);
  gemm(f_h, ppiW1bT, nullptr, bPh, nullptr, M, 320, 320, 0);
  ppi_kernel<<<dim3(M, 256 / (PPI_JT * 16)), dim3(32), 0, stream>>>(
      aPh, bPh, ppi_b1, ppi_b2, ppiW2t, ppi_W3, ppi_b3, ppi_out);

  // enzyme head: ec = softmax(relu(f@W1+b1)@W2+b2)
  gemm(f_h, enzW1t, enz_b1, e1_h, nullptr, M, 640, 320, 1);
  gemm(e1_h, enzW2t, enz_b2, nullptr, elog, M, 7, 640, 0);
  softmax7_kernel<<<dim3((M + 255) / 256), dim3(256), 0, stream>>>(elog, ec_out, M);

  // activity head: sigmoid(relu([f|ec]@W1+b1)@W2+b2)
  build_cat_kernel<<<dim3((M * 352 + 255) / 256), dim3(256), 0, stream>>>(
      f_h, ec_out, cat_h, M);
  gemm(cat_h, actW1t, act_b1, a1_h, nullptr, M, 320, 352, 1);
  matvec_sigmoid_kernel<<<dim3(M / 4), dim3(128), 0, stream>>>(
      a1_h, act_W2, act_b2, act_out, M, 320);

  // binding head: sigmoid(relu(relu(f@W1+b1)@W2+b2)@W3+b3)
  gemm(f_h, bindW1t, bind_b1, bh1, nullptr, M, 640, 320, 1);
  gemm(bh1, bindW2t, bind_b2, bh2, nullptr, M, 320, 640, 1);
  matvec_sigmoid_kernel<<<dim3(M / 4), dim3(128), 0, stream>>>(
      bh2, bind_W3, bind_b3, bind_out, M, 320);
}
